// GAT_LPA_11227044512193
// MI455X (gfx1250) — matrix operand order repaired and checked by reading, not compiled, not run on hardware
//
#include <hip/hip_runtime.h>
#include <math.h>

#define NNODES 50000
#define NEDGES 800000
#define SLOPE  0.2f

typedef __attribute__((ext_vector_type(16))) _Float16 v16h;
typedef __attribute__((ext_vector_type(8)))  _Float16 v8h;
typedef __attribute__((ext_vector_type(8)))  float    v8f;

// ---------- monotonic float <-> uint encoding for atomic max ----------
__device__ __forceinline__ unsigned f2ord(float f) {
    unsigned u = __float_as_uint(f);
    return (u & 0x80000000u) ? ~u : (u | 0x80000000u);
}
__device__ __forceinline__ float ord2f(unsigned u) {
    return (u & 0x80000000u) ? __uint_as_float(u & 0x7FFFFFFFu)
                             : __uint_as_float(~u);
}

// ---------- zero fill ----------
__global__ void zero_f32(float* __restrict__ p, int n) {
    int i = blockIdx.x * blockDim.x + threadIdx.x;
    if (i < n) p[i] = 0.0f;
}

// ---------- f32 -> f16 elementwise (layer-1 input only) ----------
__global__ void to_f16(const float* __restrict__ in, _Float16* __restrict__ out, int n) {
    int i = blockIdx.x * blockDim.x + threadIdx.x;
    if (i < n) out[i] = (_Float16)in[i];
}

// ---------- pack W[K,NC] (f32 row-major) into WMMA B-fragment layout ----------
// Bp layout: [ct = NC/16][ks = K/32][lane 0..31][16 halves], so each lane's
// fragment for one k-step is 32 contiguous bytes.
// Lane l (<16): column ct*16+l, K = ks*32 + 0..15 ; lane l+16: same col, K = ks*32 + 16..31.
__global__ void pack_w(const float* __restrict__ W, _Float16* __restrict__ Bp, int K, int NC) {
    int idx = blockIdx.x * blockDim.x + threadIdx.x;
    if (idx >= K * NC) return;
    int i    = idx & 15;
    int lane = (idx >> 4) & 31;
    int blk  = idx >> 9;            // ct*(K/32) + ks
    int ks   = blk % (K / 32);
    int ct   = blk / (K / 32);
    int col  = ct * 16 + (lane & 15);
    int k    = ks * 32 + ((lane < 16) ? 0 : 16) + i;
    Bp[idx] = (_Float16)W[(size_t)k * NC + col];
}

// ---------- WMMA GEMM: H[M,NC] = Xh[M,K] @ Wp (pre-packed), f16 math, f32 acc
// block = 128 threads (4 waves); wave w computes 16x16 tile
// (blockIdx.x*16, (blockIdx.y*4+w)*16). M is a multiple of 16 (50000 = 3125*16).
template <int K>
__global__ void gemm_wmma(const _Float16* __restrict__ Xh, const _Float16* __restrict__ Wp,
                          float* __restrict__ H, int NC) {
    const int lane = threadIdx.x & 31;
    const int wv   = threadIdx.x >> 5;
    const int row0 = blockIdx.x * 16;
    const int ct   = blockIdx.y * 4 + wv;   // 16-col tile index
    const int r    = lane & 15;
    const int akb  = (lane < 16) ? 0 : 16;  // A K-offset (halves), contiguous k = 16h + s
    const int mh   = (lane < 16) ? 0 : 8;   // C/D M offset
    constexpr int KS = K / 32;

    const _Float16* ap = Xh + (size_t)(row0 + r) * K + akb;
    const _Float16* bp = Wp + ((size_t)ct * KS * 32 + lane) * 16;

    v8f acc = {};
#pragma unroll
    for (int ks = 0; ks < KS; ++ks) {
        v8h alo = *(const v8h*)(ap + ks * 32);
        v8h ahi = *(const v8h*)(ap + ks * 32 + 8);
        v16h a = __builtin_shufflevector(alo, ahi, 0, 1, 2, 3, 4, 5, 6, 7,
                                         8, 9, 10, 11, 12, 13, 14, 15);
        v16h b = *(const v16h*)(bp + (size_t)ks * 512);
        acc = __builtin_amdgcn_wmma_f32_16x16x32_f16(
            /*neg_a=*/false, a, /*neg_b=*/false, b,
            /*c_mod=*/(short)0, acc, /*reuse_a=*/false, /*reuse_b=*/false);
    }
    float* hp = H + (size_t)(row0 + mh) * NC + ct * 16 + r;
#pragma unroll
    for (int j = 0; j < 8; ++j)
        hp[(size_t)j * NC] = acc[j];
}

// ---------- per-node attention logits: als/ald[n,h] = <H[n,h,:], a_src/dst[h,:]>
__global__ void attn_logits(const float* __restrict__ H, const float* __restrict__ a_src,
                            const float* __restrict__ a_dst, float* __restrict__ als,
                            float* __restrict__ ald, int n, int heads, int ch) {
    int idx = blockIdx.x * blockDim.x + threadIdx.x;
    if (idx >= n * heads) return;
    int node = idx / heads, h = idx % heads;
    const float* hp = H + (size_t)node * heads * ch + (size_t)h * ch;
    const float* as = a_src + h * ch;
    const float* ad = a_dst + h * ch;
    float s = 0.f, d = 0.f;
    for (int c = 0; c < ch; ++c) { float v = hp[c]; s += v * as[c]; d += v * ad[c]; }
    als[idx] = s;
    ald[idx] = d;
}

// ---------- segment max over incoming edges (ordered-uint atomic max) ----------
__global__ void seg_max(const long long* __restrict__ src, const long long* __restrict__ dst,
                        const float* __restrict__ als, const float* __restrict__ ald,
                        unsigned* __restrict__ menc, int E, int heads) {
    int idx = blockIdx.x * blockDim.x + threadIdx.x;
    if (idx >= E * heads) return;
    int e = idx / heads, h = idx % heads;
    int s = (int)src[e], d = (int)dst[e];
    float z = als[s * heads + h] + ald[d * heads + h];
    z = z > 0.f ? z : SLOPE * z;
    atomicMax(&menc[d * heads + h], f2ord(z));
}

// ---------- e = exp(logit - max) * w; denom[dst] += e ----------
__global__ void exp_sum(const long long* __restrict__ src, const long long* __restrict__ dst,
                        const float* __restrict__ ew, const float* __restrict__ als,
                        const float* __restrict__ ald, const unsigned* __restrict__ menc,
                        float* __restrict__ ebuf, float* __restrict__ den, int E, int heads) {
    int idx = blockIdx.x * blockDim.x + threadIdx.x;
    if (idx >= E * heads) return;
    int e = idx / heads, h = idx % heads;
    int s = (int)src[e], d = (int)dst[e];
    float z = als[s * heads + h] + ald[d * heads + h];
    z = z > 0.f ? z : SLOPE * z;
    float m  = ord2f(menc[d * heads + h]);
    float ev = __expf(z - m) * ew[e];
    ebuf[idx] = ev;
    atomicAdd(&den[d * heads + h], ev);
}

// ---------- wave-per-edge weighted gather/scatter: out[dst] += h[src] * alpha ----------
template <int HEADS_, int CH>
__global__ void aggregate(const long long* __restrict__ src, const long long* __restrict__ dst,
                          const float* __restrict__ H, const float* __restrict__ ebuf,
                          const float* __restrict__ den, float* __restrict__ out, int E) {
    const int HC  = HEADS_ * CH;
    const int CPL = HC / 32; // channels per lane
    int lane = threadIdx.x & 31;
    int e = (blockIdx.x * blockDim.x + threadIdx.x) >> 5;
    if (e >= E) return;
    int s = (int)src[e], d = (int)dst[e];
#pragma unroll
    for (int i = 0; i < CPL; ++i) {
        int c = lane * CPL + i;
        int h = c / CH;
        float alpha = ebuf[(size_t)e * HEADS_ + h] / (den[(size_t)d * HEADS_ + h] + 1e-16f);
        atomicAdd(&out[(size_t)d * HC + c], H[(size_t)s * HC + c] * alpha);
    }
}

// ---------- activation epilogue, f32 in-place (final layer) ----------
__global__ void finalize(float* __restrict__ io, const float* __restrict__ bias,
                         int n, int dim, int do_relu) {
    int idx = blockIdx.x * blockDim.x + threadIdx.x;
    if (idx >= n * dim) return;
    float v = io[idx] + bias[idx % dim];
    io[idx] = do_relu ? fmaxf(v, 0.f) : v;
}

// ---------- activation epilogue fused with f16 conversion (feeds next GEMM) ----------
__global__ void finalize_f16(const float* __restrict__ agg, const float* __restrict__ bias,
                             _Float16* __restrict__ outh, int n, int dim) {
    int idx = blockIdx.x * blockDim.x + threadIdx.x;
    if (idx >= n * dim) return;
    float v = agg[idx] + bias[idx % dim];
    outh[idx] = (_Float16)fmaxf(v, 0.f);
}

static inline int cdiv(int a, int b) { return (a + b - 1) / b; }

extern "C" void kernel_launch(void* const* d_in, const int* in_sizes, int n_in,
                              void* d_out, int out_size, void* d_ws, size_t ws_size,
                              hipStream_t stream) {
    const float*     x   = (const float*)d_in[0];
    const long long* ei  = (const long long*)d_in[1];
    const float*     ew  = (const float*)d_in[2];
    const float *W1 = (const float*)d_in[3],  *as1 = (const float*)d_in[4],
                *ad1 = (const float*)d_in[5], *b1  = (const float*)d_in[6];
    const float *W2 = (const float*)d_in[7],  *as2 = (const float*)d_in[8],
                *ad2 = (const float*)d_in[9], *b2  = (const float*)d_in[10];
    const float *W3 = (const float*)d_in[11], *as3 = (const float*)d_in[12],
                *ad3 = (const float*)d_in[13], *b3 = (const float*)d_in[14];
    const long long* srcI = ei;
    const long long* dstI = ei + NEDGES;

    // workspace layout (all 256B aligned)
    char* ws = (char*)d_ws;
    size_t off = 0;
    auto alloc = [&](size_t bytes) { char* p = ws + off; off = (off + bytes + 255) & ~(size_t)255; return p; };
    _Float16* Xh  = (_Float16*)alloc((size_t)NNODES * 256 * 2); // f16 activations (layer1: 256-wide)
    float*    A   = (float*)alloc((size_t)NNODES * 128 * 4);    // GEMM outputs (H)
    float*    B   = (float*)alloc((size_t)NNODES * 128 * 4);    // aggregation target
    float*    ALS = (float*)alloc((size_t)NNODES * 4 * 4);
    float*    ALD = (float*)alloc((size_t)NNODES * 4 * 4);
    unsigned* MEN = (unsigned*)alloc((size_t)NNODES * 4 * 4);
    float*    DEN = (float*)alloc((size_t)NNODES * 4 * 4);
    float*    EBF = (float*)alloc((size_t)NEDGES * 4 * 4);
    _Float16* Wp1 = (_Float16*)alloc((size_t)256 * 128 * 2);
    _Float16* Wp2 = (_Float16*)alloc((size_t)128 * 128 * 2);
    _Float16* Wp3 = (_Float16*)alloc((size_t)128 * 64 * 2);
    float*    OUT = (float*)d_out; // [NNODES, 64]

    const int TB = 256;
    const int Mtiles = NNODES / 16; // 3125
    dim3 g1(Mtiles, 2), g3(Mtiles, 1), blk(128);

    // ---------------- prep: pack weights, convert input to f16 ----------------
    pack_w<<<cdiv(256 * 128, TB), TB, 0, stream>>>(W1, Wp1, 256, 128);
    pack_w<<<cdiv(128 * 128, TB), TB, 0, stream>>>(W2, Wp2, 128, 128);
    pack_w<<<cdiv(128 * 64,  TB), TB, 0, stream>>>(W3, Wp3, 128, 64);
    to_f16<<<cdiv(NNODES * 256, TB), TB, 0, stream>>>(x, Xh, NNODES * 256);

    // ---------------- layer 1: Xh[.,256] -> 4 heads x 32 ----------------
    gemm_wmma<256><<<g1, blk, 0, stream>>>(Xh, Wp1, A, 128);
    attn_logits<<<cdiv(NNODES * 4, TB), TB, 0, stream>>>(A, as1, ad1, ALS, ALD, NNODES, 4, 32);
    zero_f32<<<cdiv(NNODES * 4, TB), TB, 0, stream>>>((float*)MEN, NNODES * 4);
    zero_f32<<<cdiv(NNODES * 4, TB), TB, 0, stream>>>(DEN, NNODES * 4);
    zero_f32<<<cdiv(NNODES * 128, TB), TB, 0, stream>>>(B, NNODES * 128);
    seg_max<<<cdiv(NEDGES * 4, TB), TB, 0, stream>>>(srcI, dstI, ALS, ALD, MEN, NEDGES, 4);
    exp_sum<<<cdiv(NEDGES * 4, TB), TB, 0, stream>>>(srcI, dstI, ew, ALS, ALD, MEN, EBF, DEN, NEDGES, 4);
    aggregate<4, 32><<<cdiv(NEDGES * 32, TB), TB, 0, stream>>>(srcI, dstI, A, EBF, DEN, B, NEDGES);
    finalize_f16<<<cdiv(NNODES * 128, TB), TB, 0, stream>>>(B, b1, Xh, NNODES, 128); // Xh = relu(agg+b1) in f16

    // ---------------- layer 2: Xh[.,128] -> 4 heads x 32 ----------------
    gemm_wmma<128><<<g1, blk, 0, stream>>>(Xh, Wp2, A, 128);
    attn_logits<<<cdiv(NNODES * 4, TB), TB, 0, stream>>>(A, as2, ad2, ALS, ALD, NNODES, 4, 32);
    zero_f32<<<cdiv(NNODES * 4, TB), TB, 0, stream>>>((float*)MEN, NNODES * 4);
    zero_f32<<<cdiv(NNODES * 4, TB), TB, 0, stream>>>(DEN, NNODES * 4);
    zero_f32<<<cdiv(NNODES * 128, TB), TB, 0, stream>>>(B, NNODES * 128);
    seg_max<<<cdiv(NEDGES * 4, TB), TB, 0, stream>>>(srcI, dstI, ALS, ALD, MEN, NEDGES, 4);
    exp_sum<<<cdiv(NEDGES * 4, TB), TB, 0, stream>>>(srcI, dstI, ew, ALS, ALD, MEN, EBF, DEN, NEDGES, 4);
    aggregate<4, 32><<<cdiv(NEDGES * 32, TB), TB, 0, stream>>>(srcI, dstI, A, EBF, DEN, B, NEDGES);
    finalize_f16<<<cdiv(NNODES * 128, TB), TB, 0, stream>>>(B, b2, Xh, NNODES, 128); // Xh = relu(agg+b2)

    // ---------------- layer 3: Xh[.,128] -> 1 head x 64 ----------------
    gemm_wmma<128><<<g3, blk, 0, stream>>>(Xh, Wp3, A, 64);
    attn_logits<<<cdiv(NNODES, TB), TB, 0, stream>>>(A, as3, ad3, ALS, ALD, NNODES, 1, 64);
    zero_f32<<<cdiv(NNODES, TB), TB, 0, stream>>>((float*)MEN, NNODES);
    zero_f32<<<cdiv(NNODES, TB), TB, 0, stream>>>(DEN, NNODES);
    zero_f32<<<cdiv(NNODES * 64, TB), TB, 0, stream>>>(OUT, NNODES * 64);
    seg_max<<<cdiv(NEDGES, TB), TB, 0, stream>>>(srcI, dstI, ALS, ALD, MEN, NEDGES, 1);
    exp_sum<<<cdiv(NEDGES, TB), TB, 0, stream>>>(srcI, dstI, ew, ALS, ALD, MEN, EBF, DEN, NEDGES, 1);
    aggregate<1, 64><<<cdiv(NEDGES * 32, TB), TB, 0, stream>>>(srcI, dstI, A, EBF, DEN, OUT, NEDGES);
    finalize<<<cdiv(NNODES * 64, TB), TB, 0, stream>>>(OUT, b3, NNODES, 64, 0); // + b3, no relu
    (void)in_sizes; (void)n_in; (void)out_size; (void)ws_size;
}
